// RNN_52939766890604
// MI455X (gfx1250) — matrix operand order repaired and checked by reading, not compiled, not run on hardware
//
#include <hip/hip_runtime.h>
#include <math.h>

// ---- sizes (fixed by the reference) ----
#define TT   512
#define BB   128
#define DIN  128
#define HH   512
#define STEP_ELEMS (BB * HH)          // 65536 floats per time step
#define NWG_RNN 32                    // persistent workgroups for recurrence

typedef __bf16 bf16_t;
typedef __attribute__((ext_vector_type(16))) __bf16 v16bf;
typedef __attribute__((ext_vector_type(8)))  __bf16 v8bf;
typedef __attribute__((ext_vector_type(8)))  float  v8f;

union V16 { v16bf v; v8bf h[2]; };

// convert 8 contiguous f32 -> 8 bf16 (register-resident)
__device__ __forceinline__ v8bf cvt8(const float* p) {
    v8f f = *(const v8f*)p;
    v8bf r;
#pragma unroll
    for (int i = 0; i < 8; ++i) r[i] = (bf16_t)f[i];
    return r;
}

// A fragment (16x32 bf16 tile) for this lane, from an f32 row.
// Lane L<16 (hi=0): elems 0..15 = K k..k+15
// Lane L>=16 (hi=1): elems 0..15 = K k+16..k+31
__device__ __forceinline__ v16bf load_a_f32(const float* row, int k, int hi) {
    V16 u;
    u.h[0] = cvt8(row + k + 16 * hi);
    u.h[1] = cvt8(row + k + 16 * hi + 8);
    return u.v;
}

__device__ __forceinline__ v16bf load_a_bf16(const bf16_t* row, int k, int hi) {
    V16 u;
    u.h[0] = *(const v8bf*)(row + k + 16 * hi);
    u.h[1] = *(const v8bf*)(row + k + 16 * hi + 8);
    return u.v;
}

// ---------------- init: zero h ping-pong buffers + barrier words ----------------
__global__ void __launch_bounds__(256) rnn_init_kernel(unsigned* hbufs_u32, unsigned* bar) {
    int i = blockIdx.x * 256 + threadIdx.x;        // 64 blocks * 256 = 16384 threads
    if (i < 2) bar[i] = 0u;
    for (int j = i; j < 2 * STEP_ELEMS / 2; j += 64 * 256)   // 2 * 65536 bf16 = 65536 u32
        hbufs_u32[j] = 0u;
}

// ---------------- phase 1: xp = x @ W_ih^T + b_ih  (written into d_out h_seq) ----------------
// M = T*B = 65536, N = H = 512, K = D_in = 128. One 16x16 tile per wave.
__global__ void __launch_bounds__(256) xproj_kernel(const float* __restrict__ x,
                                                    const float* __restrict__ W_ih,
                                                    const float* __restrict__ b_ih,
                                                    float* __restrict__ out) {
    const int wid  = blockIdx.x * 8 + (threadIdx.x >> 5);   // global wave id
    const int lane = threadIdx.x & 31;
    const int hi = lane >> 4, lo = lane & 15;
    const int nb = wid & 31;        // 32 N-tiles
    const int mb = wid >> 5;        // 4096 M-tiles

    const float* arow = x    + (size_t)(mb * 16 + lo) * DIN;            // A: row of x
    const float* brow = W_ih + (size_t)(nb * 16 + lo) * DIN + 16 * hi;  // B: row of W_ih

    v8f c = {};
#pragma unroll
    for (int k = 0; k < DIN; k += 32) {
        v16bf a = load_a_f32(arow, k, hi);
        V16 ub;
        ub.h[0] = cvt8(brow + k);
        ub.h[1] = cvt8(brow + k + 8);
        c = __builtin_amdgcn_wmma_f32_16x16x32_bf16(false, a, false, ub.v,
                                                    (short)0, c, false, false);
    }

    const int   gcol = nb * 16 + lo;
    const float bias = b_ih[gcol];
    const size_t base = (size_t)(mb * 16 + 8 * hi) * HH + gcol;
#pragma unroll
    for (int v = 0; v < 8; ++v)
        out[base + (size_t)v * HH] = c[v] + bias;
}

// ---------------- grid-wide sense barrier (persistent kernel) ----------------
__device__ __forceinline__ void grid_sync(unsigned* cnt, unsigned* gen) {
    __syncthreads();
    if (threadIdx.x == 0) {
        __threadfence();
        unsigned g = __hip_atomic_load(gen, __ATOMIC_RELAXED, __HIP_MEMORY_SCOPE_AGENT);
        unsigned a = __hip_atomic_fetch_add(cnt, 1u, __ATOMIC_ACQ_REL, __HIP_MEMORY_SCOPE_AGENT);
        if (a == NWG_RNN - 1) {
            __hip_atomic_store(cnt, 0u, __ATOMIC_RELAXED, __HIP_MEMORY_SCOPE_AGENT);
            __hip_atomic_fetch_add(gen, 1u, __ATOMIC_ACQ_REL, __HIP_MEMORY_SCOPE_AGENT);
        } else {
            while (__hip_atomic_load(gen, __ATOMIC_ACQUIRE, __HIP_MEMORY_SCOPE_AGENT) == g)
                __builtin_amdgcn_s_sleep(1);
        }
        __threadfence();
    }
    __syncthreads();
}

// ---------------- phase 2: persistent recurrence ----------------
// 32 WGs x 8 waves; WG nb owns output columns [16*nb, 16*nb+16), wave mb owns rows
// [16*mb, 16*mb+16). W_hh slice cached in LDS (bf16). h double-buffered in ws (bf16).
__global__ void __launch_bounds__(256) rnn_steps_kernel(const float* __restrict__ W_hh,
                                                        const float* __restrict__ b_hh,
                                                        float* out,
                                                        bf16_t* hb0, bf16_t* hb1,
                                                        unsigned* bar) {
    __shared__ __align__(32) bf16_t wtile[16 * HH];   // 16 KB

    const int nb  = blockIdx.x;
    const int tid = threadIdx.x;

    // stage this WG's 16 rows of W_hh into LDS as bf16 (once)
    {
        const float* src = W_hh + (size_t)nb * 16 * HH;
        for (int i = tid; i < 16 * HH; i += 256)
            wtile[i] = (bf16_t)src[i];
    }
    __syncthreads();

    const int mb = tid >> 5;
    const int lane = tid & 31;
    const int hi = lane >> 4, lo = lane & 15;

    const int    gcol   = nb * 16 + lo;
    const float  bias   = b_hh[gcol];
    const size_t rowoff = (size_t)(mb * 16 + lo) * HH;                 // A row offset in h
    const size_t base   = (size_t)(mb * 16 + 8 * hi) * HH + gcol;      // C base in (B,H)
    const bf16_t* brow  = wtile + (size_t)lo * HH + 16 * hi;           // B row in LDS

    for (int t = 0; t < TT; ++t) {
        const bf16_t* hcur = (t & 1) ? hb1 : hb0;   // read buffer
        bf16_t*       hnxt = (t & 1) ? hb0 : hb1;   // write buffer
        const bf16_t* arow = hcur + rowoff;

        float* xp = out + (size_t)t * STEP_ELEMS;   // xp[t] lives in d_out; overwritten below
        if (t + 1 < TT)                              // gfx1250 global_prefetch_b8
            __builtin_prefetch((const void*)(out + (size_t)(t + 1) * STEP_ELEMS + base), 0, 1);

        v8f c = {};
#pragma unroll
        for (int k = 0; k < HH; k += 32) {
            v16bf a = load_a_bf16(arow, k, hi);
            v16bf b = *(const v16bf*)(brow + k);
            c = __builtin_amdgcn_wmma_f32_16x16x32_bf16(false, a, false, b,
                                                        (short)0, c, false, false);
        }

#pragma unroll
        for (int v = 0; v < 8; ++v) {
            const size_t off = base + (size_t)v * HH;
            float val = tanhf(xp[off] + c[v] + bias);
            xp[off] = val;                    // h_seq[t] overwrites xp[t] in place
            hnxt[off] = (bf16_t)val;          // bf16 h for next step's GEMM
            if (t == TT - 1)
                out[(size_t)TT * STEP_ELEMS + off] = val;   // h_last
        }

        grid_sync(bar, bar + 1);
    }
}

extern "C" void kernel_launch(void* const* d_in, const int* in_sizes, int n_in,
                              void* d_out, int out_size, void* d_ws, size_t ws_size,
                              hipStream_t stream) {
    const float* x    = (const float*)d_in[0];
    const float* W_ih = (const float*)d_in[1];
    const float* b_ih = (const float*)d_in[2];
    const float* W_hh = (const float*)d_in[3];
    const float* b_hh = (const float*)d_in[4];
    float* out = (float*)d_out;

    // ws layout: hb0 (128KB bf16) | hb1 (128KB bf16) | barrier (2 x u32)
    bf16_t*   hb0 = (bf16_t*)d_ws;
    bf16_t*   hb1 = (bf16_t*)((char*)d_ws + (size_t)STEP_ELEMS * sizeof(bf16_t));
    unsigned* bar = (unsigned*)((char*)d_ws + 2 * (size_t)STEP_ELEMS * sizeof(bf16_t));

    rnn_init_kernel<<<64, 256, 0, stream>>>((unsigned*)d_ws, bar);
    xproj_kernel<<<(TT * BB / 16) * (HH / 16) / 8, 256, 0, stream>>>(x, W_ih, b_ih, out);
    rnn_steps_kernel<<<NWG_RNN, 256, 0, stream>>>(W_hh, b_hh, out, hb0, hb1, bar);
}
